// SeqAttnMatch_56796647522731
// MI455X (gfx1250) — matrix operand order repaired and checked by reading, not compiled, not run on hardware
//
#include <hip/hip_runtime.h>
#include <stdint.h>

// Problem constants (match reference)
#define B_  8
#define L1_ 2048
#define L2_ 2048
#define H_  1024

typedef __attribute__((ext_vector_type(16))) __bf16        v16bf;
typedef __attribute__((ext_vector_type(8)))  float         v8f;
typedef __attribute__((ext_vector_type(4)))  unsigned int  u32x4;

union FragBF { u32x4 q[2]; v16bf v; };

__device__ __forceinline__ unsigned short f2bf(float f) {
  union { float f; unsigned int u; } c;
  c.f = f;
  unsigned int u = c.u;
  unsigned int r = u + 0x7FFFu + ((u >> 16) & 1u);   // round-to-nearest-even
  return (unsigned short)(r >> 16);
}

__device__ __forceinline__ v8f vzero8() {
  v8f z;
#pragma unroll
  for (int i = 0; i < 8; ++i) z[i] = 0.0f;
  return z;
}

// ---------------------------------------------------------------------------
// Kernel 1: f32 -> bf16 elementwise (weights)
// ---------------------------------------------------------------------------
__global__ void cvt_bf16_kernel(const float* __restrict__ in,
                                unsigned short* __restrict__ out, int n) {
  int i = blockIdx.x * blockDim.x + threadIdx.x;
  if (i < n) out[i] = f2bf(in[i]);
}

// ---------------------------------------------------------------------------
// Kernel 2: y[B,L2,H] f32 -> YbT[B,H,L2] bf16 (tiled transpose via LDS)
// ---------------------------------------------------------------------------
__global__ void transpose_y_bf16_kernel(const float* __restrict__ y,
                                        unsigned short* __restrict__ yT) {
  __shared__ unsigned short tile[16][17];
  int blk = blockIdx.x;
  int b   = blk >> 13;          // / (128*64)
  int rem = blk & 8191;
  int jt  = rem >> 6;           // 128 j-tiles
  int ht  = rem & 63;           // 64 h-tiles
  int tx  = threadIdx.x & 15, ty = threadIdx.x >> 4;
  int j0  = jt * 16, h0 = ht * 16;
  tile[ty][tx] = f2bf(y[((size_t)(b * L2_ + j0 + ty)) * H_ + h0 + tx]);
  __syncthreads();
  yT[((size_t)(b * H_ + h0 + ty)) * L2_ + j0 + tx] = tile[tx][ty];
}

// ---------------------------------------------------------------------------
// Kernel 3: proj = relu(src @ W^T + bias), bf16 out.
// One block = (batch, 16-row tile). 8 waves, each owns 128 output columns.
// ---------------------------------------------------------------------------
__global__ __launch_bounds__(256) void proj_relu_bf16_kernel(
    const float* __restrict__ src,           // [B, L, H] f32
    const unsigned short* __restrict__ Wb,   // [H, H] bf16, row = out feature
    const float* __restrict__ bias,          // [H] f32
    unsigned short* __restrict__ dst,        // [B, L, H] bf16
    int L) {
  __shared__ unsigned short atile[16 * 1040];   // 16 x H bf16, padded rows

  int blk    = blockIdx.x;
  int mtiles = L >> 4;
  int b      = blk / mtiles;
  int row0   = (blk % mtiles) << 4;
  int tid    = threadIdx.x;

  // stage 16 source rows into LDS as bf16
  for (int e = tid; e < 16 * H_; e += 256) {
    int r = e >> 10, c = e & (H_ - 1);
    atile[r * 1040 + c] = f2bf(src[((size_t)(b * L + row0 + r)) * H_ + c]);
  }
  __syncthreads();

  int wave = tid >> 5, lane = tid & 31;
  int ln = lane & 15, hi = lane >> 4;
  int hbase = wave * 128;

  v8f acc[8];
#pragma unroll
  for (int t = 0; t < 8; ++t) acc[t] = vzero8();

  const unsigned short* arow = atile + ln * 1040;
  for (int k = 0; k < H_; k += 32) {
    FragBF a;
    a.q[0] = *(const u32x4*)(arow + k + hi * 8);
    a.q[1] = *(const u32x4*)(arow + k + 16 + hi * 8);
#pragma unroll
    for (int t = 0; t < 8; ++t) {
      int h = hbase + t * 16 + ln;
      const unsigned short* bp = Wb + (size_t)h * H_ + k + hi * 8;
      FragBF bb;
      bb.q[0] = *(const u32x4*)bp;
      bb.q[1] = *(const u32x4*)(bp + 16);
      acc[t] = __builtin_amdgcn_wmma_f32_16x16x32_bf16(
          false, a.v, false, bb.v, (short)0, acc[t], false, false);
    }
  }

#pragma unroll
  for (int t = 0; t < 8; ++t) {
    int h = hbase + t * 16 + ln;
    float bv = bias[h];
#pragma unroll
    for (int r = 0; r < 8; ++r) {
      int m = r + hi * 8;
      float v = acc[t][r] + bv;
      v = v > 0.0f ? v : 0.0f;
      dst[((size_t)(b * L + row0 + m)) * H_ + h] = f2bf(v);
    }
  }
}

// ---------------------------------------------------------------------------
// Kernel 4: fused attention for one (batch, 16 query rows) tile.
//   Phase 1: S = Xp_tile @ Yp^T  (each wave: 16x256 slab in registers)
//   masked softmax (shfl + LDS reductions), P stored bf16 in LDS
//   Phase 2: out = (P @ Y) / rowsum  (each wave: 128 output columns)
// ---------------------------------------------------------------------------
#define PSTRIDE 2064
__global__ __launch_bounds__(256) void attn_kernel(
    const unsigned short* __restrict__ Xp,   // [B,L1,H] bf16
    const unsigned short* __restrict__ Yp,   // [B,L2,H] bf16
    const unsigned short* __restrict__ YbT,  // [B,H,L2] bf16
    const unsigned char* __restrict__ mask,  // [B,L2] bool
    float* __restrict__ out) {               // [B,L1,H] f32
  __shared__ unsigned short Pld[16 * PSTRIDE];   // probabilities bf16
  __shared__ unsigned char  mld[L2_];
  __shared__ float rowmax[16];
  __shared__ float rowsum[16];
  __shared__ float wred[8][16];

  int blk  = blockIdx.x;
  int b    = blk >> 7;            // L1/16 == 128 tiles per batch
  int row0 = (blk & 127) << 4;
  int tid  = threadIdx.x;

  for (int e = tid; e < L2_; e += 256) mld[e] = mask[b * L2_ + e];
  __syncthreads();

  int wave = tid >> 5, lane = tid & 31;
  int ln = lane & 15, hi = lane >> 4;

  // -------- Phase 1: score slab for j in [jbase, jbase+256) --------
  int jbase = wave * 256;
  v8f S[16];
#pragma unroll
  for (int t = 0; t < 16; ++t) S[t] = vzero8();

  const unsigned short* arow = Xp + ((size_t)(b * L1_ + row0 + ln)) * H_;
  for (int k = 0; k < H_; k += 32) {
    FragBF a;
    a.q[0] = *(const u32x4*)(arow + k + hi * 8);
    a.q[1] = *(const u32x4*)(arow + k + 16 + hi * 8);
#pragma unroll
    for (int jt = 0; jt < 16; ++jt) {
      int j = jbase + jt * 16 + ln;
      const unsigned short* bp = Yp + ((size_t)(b * L2_ + j)) * H_ + k + hi * 8;
      FragBF bb;
      bb.q[0] = *(const u32x4*)bp;
      bb.q[1] = *(const u32x4*)(bp + 16);
      S[jt] = __builtin_amdgcn_wmma_f32_16x16x32_bf16(
          false, a.v, false, bb.v, (short)0, S[jt], false, false);
    }
  }

  // mask + per-lane row maxima (lane holds rows hi*8 .. hi*8+7)
  float lmax[8];
#pragma unroll
  for (int r = 0; r < 8; ++r) lmax[r] = -3.0e38f;
#pragma unroll
  for (int jt = 0; jt < 16; ++jt) {
    int j = jbase + jt * 16 + ln;
    bool msk = mld[j] != 0;
#pragma unroll
    for (int r = 0; r < 8; ++r) {
      float s = msk ? -3.0e38f : S[jt][r];
      S[jt][r] = s;
      lmax[r]  = fmaxf(lmax[r], s);
    }
  }
  // reduce over the 16 lanes sharing the same hi-half
#pragma unroll
  for (int off = 1; off < 16; off <<= 1) {
#pragma unroll
    for (int r = 0; r < 8; ++r)
      lmax[r] = fmaxf(lmax[r], __shfl_xor(lmax[r], off, 32));
  }
  if (ln == 0) {
#pragma unroll
    for (int r = 0; r < 8; ++r) wred[wave][hi * 8 + r] = lmax[r];
  }
  __syncthreads();
  if (tid < 16) {
    float mx = wred[0][tid];
    for (int w = 1; w < 8; ++w) mx = fmaxf(mx, wred[w][tid]);
    rowmax[tid] = mx;
  }
  __syncthreads();

  // exp, write P (bf16) into LDS, accumulate row sums
  float rmx[8];
#pragma unroll
  for (int r = 0; r < 8; ++r) rmx[r] = rowmax[hi * 8 + r];
  float lsum[8];
#pragma unroll
  for (int r = 0; r < 8; ++r) lsum[r] = 0.0f;
#pragma unroll
  for (int jt = 0; jt < 16; ++jt) {
    int j = jbase + jt * 16 + ln;
#pragma unroll
    for (int r = 0; r < 8; ++r) {
      int m = hi * 8 + r;
      float p = __expf(S[jt][r] - rmx[r]);   // masked -> exp(-inf) = 0
      lsum[r] += p;
      Pld[m * PSTRIDE + j] = f2bf(p);
    }
  }
#pragma unroll
  for (int off = 1; off < 16; off <<= 1) {
#pragma unroll
    for (int r = 0; r < 8; ++r) lsum[r] += __shfl_xor(lsum[r], off, 32);
  }
  if (ln == 0) {
#pragma unroll
    for (int r = 0; r < 8; ++r) wred[wave][hi * 8 + r] = lsum[r];
  }
  __syncthreads();
  if (tid < 16) {
    float s = 0.0f;
    for (int w = 0; w < 8; ++w) s += wred[w][tid];
    rowsum[tid] = s;
  }
  __syncthreads();

  // -------- Phase 2: out_tile = P @ Y over this wave's 128 columns --------
  int hbase = wave * 128;
  v8f acc[8];
#pragma unroll
  for (int t = 0; t < 8; ++t) acc[t] = vzero8();

  const unsigned short* prow = Pld + ln * PSTRIDE;
  for (int k = 0; k < L2_; k += 32) {
    FragBF a;
    a.q[0] = *(const u32x4*)(prow + k + hi * 8);
    a.q[1] = *(const u32x4*)(prow + k + 16 + hi * 8);
#pragma unroll
    for (int t = 0; t < 8; ++t) {
      int h = hbase + t * 16 + ln;
      const unsigned short* bp = YbT + ((size_t)(b * H_ + h)) * L2_ + k + hi * 8;
      FragBF bb;
      bb.q[0] = *(const u32x4*)bp;
      bb.q[1] = *(const u32x4*)(bp + 16);
      acc[t] = __builtin_amdgcn_wmma_f32_16x16x32_bf16(
          false, a.v, false, bb.v, (short)0, acc[t], false, false);
    }
  }

  float rs[8];
#pragma unroll
  for (int r = 0; r < 8; ++r) rs[r] = rowsum[hi * 8 + r];
#pragma unroll
  for (int t = 0; t < 8; ++t) {
    int h = hbase + t * 16 + ln;
#pragma unroll
    for (int r = 0; r < 8; ++r) {
      int m = hi * 8 + r;
      out[((size_t)(b * L1_ + row0 + m)) * H_ + h] = acc[t][r] / rs[r];
    }
  }
}

// ---------------------------------------------------------------------------
// Launcher
// ---------------------------------------------------------------------------
extern "C" void kernel_launch(void* const* d_in, const int* in_sizes, int n_in,
                              void* d_out, int out_size, void* d_ws, size_t ws_size,
                              hipStream_t stream) {
  const float* x          = (const float*)d_in[0];
  const float* y          = (const float*)d_in[1];
  const float* W1         = (const float*)d_in[2];
  const float* b1         = (const float*)d_in[3];
  const float* W2         = (const float*)d_in[4];
  const float* b2         = (const float*)d_in[5];
  const unsigned char* ym = (const unsigned char*)d_in[6];
  float* out              = (float*)d_out;

  // workspace layout (bytes):
  //   W1b  @ 0        : 2 MB   (H*H bf16)
  //   W2b  @ 2 MB     : 2 MB
  //   YbT  @ 4 MB     : 32 MB  (B*H*L2 bf16)
  //   Xp   @ 36 MB    : 32 MB  (B*L1*H bf16)
  //   Yp   @ 68 MB    : 32 MB  (B*L2*H bf16)
  char* ws = (char*)d_ws;
  unsigned short* W1b = (unsigned short*)(ws);
  unsigned short* W2b = (unsigned short*)(ws + (size_t)2 * 1024 * 1024);
  unsigned short* YbT = (unsigned short*)(ws + (size_t)4 * 1024 * 1024);
  unsigned short* Xp  = (unsigned short*)(ws + (size_t)36 * 1024 * 1024);
  unsigned short* Yp  = (unsigned short*)(ws + (size_t)68 * 1024 * 1024);

  cvt_bf16_kernel<<<(H_ * H_) / 256, 256, 0, stream>>>(W1, W1b, H_ * H_);
  cvt_bf16_kernel<<<(H_ * H_) / 256, 256, 0, stream>>>(W2, W2b, H_ * H_);
  transpose_y_bf16_kernel<<<B_ * (L2_ / 16) * (H_ / 16), 256, 0, stream>>>(y, YbT);
  proj_relu_bf16_kernel<<<B_ * (L1_ / 16), 256, 0, stream>>>(x, W1b, b1, Xp, L1_);
  proj_relu_bf16_kernel<<<B_ * (L2_ / 16), 256, 0, stream>>>(y, W2b, b2, Yp, L2_);
  attn_kernel<<<B_ * (L1_ / 16), 256, 0, stream>>>(Xp, Yp, YbT, ym, out);
}
